// Hete_self_69836168233641
// MI455X (gfx1250) — matrix operand order repaired and checked by reading, not compiled, not run on hardware
//
#include <hip/hip_runtime.h>
#include <hip/hip_bf16.h>
#include <math.h>

#define M_PATHS 8
#define N_NODES 100000
#define D_DIM   256
#define H_DIM   256
#define EPS     1e-8f

typedef __attribute__((ext_vector_type(16))) __bf16 v16bf;
typedef __attribute__((ext_vector_type(8)))  float  v8f;

union Frag {
    unsigned int u[8];
    uint4        q[2];
    v16bf        v;
};

// hardware tanh if the gfx1250 builtin exists, else libm fallback
#if defined(__has_builtin) && __has_builtin(__builtin_amdgcn_tanhf)
#define FAST_TANH(x) __builtin_amdgcn_tanhf(x)
#else
#define FAST_TANH(x) tanhf(x)
#endif

// round-to-nearest-even fp32 -> bf16 (as uint16 in low bits)
__device__ __forceinline__ unsigned int bf_round(float f) {
    unsigned int u = __float_as_uint(f);
    return (u + 0x7FFFu + ((u >> 16) & 1u)) >> 16;
}
__device__ __forceinline__ unsigned int pk_bf16(float lo, float hi) {
    return (bf_round(lo) & 0xFFFFu) | (bf_round(hi) << 16);
}

// LDS layout strides (padded to dodge bank conflicts)
#define WSTRIDE 132   // uints per W row (128 data + 4 pad)
#define TSTRIDE 260   // floats per target row (256 data + 4 pad)

// ---------------------------------------------------------------------------
// Kernel 1: fused  tanh(homo @ W^T + b) -> cosine numerator/norms -> score
// grid.x = N/16 tiles, 256 threads (8 waves, wave w handles metapath m = w)
// ---------------------------------------------------------------------------
__global__ void __launch_bounds__(256)
score_wmma_kernel(const float* __restrict__ homo,      // [M, N, D]
                  const float* __restrict__ W,         // [H, D]
                  const float* __restrict__ bias,      // [H]
                  const float* __restrict__ out_emb,   // [NODE_NUM, H]
                  const int*   __restrict__ nodes,     // [N]
                  float*       __restrict__ score)     // [M, N] (workspace)
{
    extern __shared__ float smem_f[];
    unsigned int* sW  = (unsigned int*)smem_f;               // [H][WSTRIDE] packed bf16 pairs
    float*        sT  = (float*)(sW + H_DIM * WSTRIDE);      // [16][TSTRIDE] target rows
    float*        sB  = sT + 16 * TSTRIDE;                   // [H] bias
    float*        sTn = sB + H_DIM;                          // [16] target norms

    const int tid = threadIdx.x;
    const int n0  = blockIdx.x * 16;

    // ---- stage W as packed bf16 (coalesced float4 reads -> 2 packed uints) ----
    for (int idx = tid; idx < H_DIM * (D_DIM / 4); idx += 256) {
        const int h  = idx >> 6;                 // / (D/4)
        const int j2 = (idx & 63) * 2;           // uint pair index within row
        const float4 f = *(const float4*)(W + (size_t)h * D_DIM + 2 * j2);
        sW[h * WSTRIDE + j2]     = pk_bf16(f.x, f.y);
        sW[h * WSTRIDE + j2 + 1] = pk_bf16(f.z, f.w);
    }
    // ---- stage bias ----
    sB[tid] = bias[tid];
    // ---- stage 16 gathered target rows (float4) ----
    for (int idx = tid; idx < 16 * (H_DIM / 4); idx += 256) {
        const int r  = idx >> 6;
        const int h4 = (idx & 63) * 4;
        const int node = nodes[n0 + r];
        const float4 f = *(const float4*)(out_emb + (size_t)node * H_DIM + h4);
        float* dst = sT + r * TSTRIDE + h4;
        dst[0] = f.x; dst[1] = f.y; dst[2] = f.z; dst[3] = f.w;
    }
    __syncthreads();

    // ---- target norms ----
    if (tid < 16) {
        float s = 0.f;
        #pragma unroll 8
        for (int h = 0; h < H_DIM; ++h) {
            const float t = sT[tid * TSTRIDE + h];
            s = fmaf(t, t, s);
        }
        sTn[tid] = fmaxf(sqrtf(s), EPS);
    }
    __syncthreads();

    const int wave = tid >> 5;           // = metapath m
    const int lane = tid & 31;
    const int m    = wave;
    const int half = lane >> 4;          // 0: lanes 0-15, 1: lanes 16-31
    const int lrow = lane & 15;

    // ---- load A tile: 16 rows x 256 cols of homo[m], bf16-packed in regs ----
    const float* Abase = homo + ((size_t)m * N_NODES + (n0 + lrow)) * D_DIM;
    __builtin_prefetch(Abase, 0, 1);
    const int sub = half * 8;            // K sub-offset per lane half
    Frag a[8];
    #pragma unroll
    for (int kb = 0; kb < 8; ++kb) {
        const float* p = Abase + kb * 32;
        const float4 f0 = *(const float4*)(p + sub);
        const float4 f1 = *(const float4*)(p + sub + 4);
        const float4 f2 = *(const float4*)(p + 16 + sub);
        const float4 f3 = *(const float4*)(p + 16 + sub + 4);
        a[kb].u[0] = pk_bf16(f0.x, f0.y);
        a[kb].u[1] = pk_bf16(f0.z, f0.w);
        a[kb].u[2] = pk_bf16(f1.x, f1.y);
        a[kb].u[3] = pk_bf16(f1.z, f1.w);
        a[kb].u[4] = pk_bf16(f2.x, f2.y);
        a[kb].u[5] = pk_bf16(f2.z, f2.w);
        a[kb].u[6] = pk_bf16(f3.x, f3.y);
        a[kb].u[7] = pk_bf16(f3.z, f3.w);
    }

    float pn[8], ps[8];
    #pragma unroll
    for (int v = 0; v < 8; ++v) { pn[v] = 0.f; ps[v] = 0.f; }

    // ---- 16 h-tiles x 8 k-steps of v_wmma_f32_16x16x32_bf16 ----
    for (int ht = 0; ht < 16; ++ht) {
        v8f c = {};
        const int hcol  = ht * 16 + lrow;
        const int bbase = hcol * WSTRIDE + half * 4;   // + kb*16 below
        #pragma unroll
        for (int kb = 0; kb < 8; ++kb) {
            Frag b;
            const uint4* q = (const uint4*)(sW + bbase + kb * 16);
            b.q[0] = q[0];
            b.q[1] = q[2];
            c = __builtin_amdgcn_wmma_f32_16x16x32_bf16(
                    false, a[kb].v, false, b.v, (short)0, c, false, false);
        }
        const float bb = sB[hcol];
        #pragma unroll
        for (int v = 0; v < 8; ++v) {
            const float hval = FAST_TANH(c[v] + bb);
            const int   r    = v + half * 8;           // row within the 16-tile
            const float t    = sT[r * TSTRIDE + hcol];
            pn[v] = fmaf(hval, t, pn[v]);
            ps[v] = fmaf(hval, hval, ps[v]);
        }
    }

    // ---- reduce over the 16 h-columns held across lanes (within 16-lane group) ----
    #pragma unroll
    for (int off = 8; off >= 1; off >>= 1) {
        #pragma unroll
        for (int v = 0; v < 8; ++v) {
            pn[v] += __shfl_xor(pn[v], off, 16);
            ps[v] += __shfl_xor(ps[v], off, 16);
        }
    }

    if (lrow == 0) {
        #pragma unroll
        for (int v = 0; v < 8; ++v) {
            const int   r  = v + half * 8;
            const float hn = fmaxf(sqrtf(ps[v]), EPS);
            const float s  = pn[v] / (hn * sTn[r]);
            score[(size_t)m * N_NODES + (n0 + r)] = s;
        }
    }
}

// ---------------------------------------------------------------------------
// Kernel 2: softmax over the 8 metapaths + weighted sum of homo -> final[N,D]
// block = 256 threads handling 4 n's; each thread owns a float4 of D
// ---------------------------------------------------------------------------
__global__ void __launch_bounds__(256)
finalize_kernel(const float* __restrict__ homo,    // [M, N, D]
                const float* __restrict__ score,   // [M, N]
                float*       __restrict__ out)     // [N, D]
{
    const int tid = threadIdx.x;
    const int n   = blockIdx.x * 4 + (tid >> 6);   // 4 n's per block
    const int d4  = (tid & 63) * 4;

    float x[M_PATHS];
    float mx = -INFINITY;
    #pragma unroll
    for (int m = 0; m < M_PATHS; ++m) {
        x[m] = score[(size_t)m * N_NODES + n];
        mx = fmaxf(mx, x[m]);
    }
    float sum = 0.f;
    #pragma unroll
    for (int m = 0; m < M_PATHS; ++m) {
        x[m] = __expf(x[m] - mx);
        sum += x[m];
    }
    const float inv = 1.f / sum;

    float4 acc = make_float4(0.f, 0.f, 0.f, 0.f);
    #pragma unroll
    for (int m = 0; m < M_PATHS; ++m) {
        const float  w = x[m] * inv;
        const float4 h = *(const float4*)(homo + ((size_t)m * N_NODES + n) * D_DIM + d4);
        acc.x = fmaf(w, h.x, acc.x);
        acc.y = fmaf(w, h.y, acc.y);
        acc.z = fmaf(w, h.z, acc.z);
        acc.w = fmaf(w, h.w, acc.w);
    }
    *(float4*)(out + (size_t)n * D_DIM + d4) = acc;
}

extern "C" void kernel_launch(void* const* d_in, const int* in_sizes, int n_in,
                              void* d_out, int out_size, void* d_ws, size_t ws_size,
                              hipStream_t stream) {
    // setup_inputs order: nodes, homo_embeddings, W, b, out_embedding
    const int*   nodes   = (const int*)  d_in[0];
    const float* homo    = (const float*)d_in[1];
    const float* W       = (const float*)d_in[2];
    const float* bias    = (const float*)d_in[3];
    const float* out_emb = (const float*)d_in[4];
    float*       out     = (float*)d_out;
    float*       score   = (float*)d_ws;           // M*N fp32 = 3.2 MB

    const size_t shmem = (size_t)(H_DIM * WSTRIDE) * sizeof(unsigned int)
                       + (size_t)(16 * TSTRIDE) * sizeof(float)
                       + (size_t)H_DIM * sizeof(float)
                       + (size_t)16 * sizeof(float);

    score_wmma_kernel<<<dim3(N_NODES / 16), dim3(256), shmem, stream>>>(
        homo, W, bias, out_emb, nodes, score);

    finalize_kernel<<<dim3(N_NODES / 4), dim3(256), 0, stream>>>(homo, score, out);
}
